// Net_17781164606160
// MI455X (gfx1250) — matrix operand order repaired and checked by reading, not compiled, not run on hardware
//
#include <hip/hip_runtime.h>
#include <hip/hip_bf16.h>
#include <math.h>

typedef __attribute__((ext_vector_type(16))) _Float16 v16h;
typedef __attribute__((ext_vector_type(8)))  _Float16 v8h;
typedef __attribute__((ext_vector_type(8)))  float    v8f;

#define LATC 1536
#define NB   8
#define NN   32
#define NEDGE 1024          // NN*NN
#define MROWS 8192          // NB*NEDGE
#define NNODE 256           // NB*NN

// ---------------------------------------------------------------------------
// Kernel 1: bilinear grid-sample (align_corners=False, border clamp) for
// 256 node points and 8192 edge midpoints -> f16 rows of length 1536.
// ---------------------------------------------------------------------------
__global__ void align_kernel(const float* __restrict__ f0, const float* __restrict__ f1,
                             const float* __restrict__ f2, const float* __restrict__ f3,
                             const float* __restrict__ f4, const float* __restrict__ f5,
                             const float* __restrict__ P,
                             _Float16* __restrict__ Fn, _Float16* __restrict__ Ee)
{
    const float* feats[6] = {f0, f1, f2, f3, f4, f5};
    const int CH[6]  = {64, 64, 128, 256, 512, 512};
    const int HSz[6] = {128, 64, 32, 16, 8, 1};
    const int OFF[6] = {0, 64, 128, 256, 512, 1024};

    int pid = blockIdx.x;                 // 0..(256+8192-1)
    float px, py;
    int b;
    _Float16* dst;
    if (pid < NNODE) {
        b = pid >> 5;
        int i = pid & 31;
        px = P[(b * NN + i) * 2 + 0];
        py = P[(b * NN + i) * 2 + 1];
        dst = Fn + (long long)pid * LATC;
    } else {
        int q = pid - NNODE;
        b = q >> 10;
        int e = q & 1023, i = e >> 5, j = e & 31;
        px = 0.5f * (P[(b * NN + i) * 2 + 0] + P[(b * NN + j) * 2 + 0]);
        py = 0.5f * (P[(b * NN + i) * 2 + 1] + P[(b * NN + j) * 2 + 1]);
        dst = Ee + (long long)q * LATC;
    }

    for (int l = 0; l < 6; ++l) {
        int W = HSz[l], C = CH[l];
        float gx = 2.f * px / 256.f - 1.f;
        float gy = 2.f * py / 256.f - 1.f;
        float x = fminf(fmaxf(((gx + 1.f) * (float)W - 1.f) * 0.5f, 0.f), (float)(W - 1));
        float y = fminf(fmaxf(((gy + 1.f) * (float)W - 1.f) * 0.5f, 0.f), (float)(W - 1));
        float x0f = floorf(x), y0f = floorf(y);
        float wx = x - x0f, wy = y - y0f;
        int x0 = (int)x0f, y0 = (int)y0f;
        int x1 = (x0 + 1 < W) ? x0 + 1 : W - 1;
        int y1 = (y0 + 1 < W) ? y0 + 1 : W - 1;
        const float* fb = feats[l] + (long long)b * C * W * W;
        for (int c = threadIdx.x; c < C; c += blockDim.x) {
            const float* fc = fb + (long long)c * W * W;
            float v00 = fc[y0 * W + x0], v01 = fc[y0 * W + x1];
            float v10 = fc[y1 * W + x0], v11 = fc[y1 * W + x1];
            float v = v00 * (1.f - wx) * (1.f - wy) + v01 * wx * (1.f - wy)
                    + v10 * (1.f - wx) * wy + v11 * wx * wy;
            dst[OFF[l] + c] = (_Float16)v;
        }
    }
}

// ---------------------------------------------------------------------------
// Kernel 2: weight repack (fp32 -> f16, [out_ch][K] layout) + BN folding.
//   Wnode  [2048][1536]: rows 0..511 g_W1, 512..1023 g_W2, 1024..1535 p_W1, 1536..2047 p_W2
//   Wintro [1024][1536]: rows 0..511 g_W3, 512..1023 p_W3   (E slice of iw)
//   Wcw    [1024][512] : rows 0..511 g_cw, 512..1023 p_cw
//   sbn/tbn: BN folded scale/shift; biasc = intro bias; cbc = chorus bias
// ---------------------------------------------------------------------------
__global__ void prep_kernel(const float* __restrict__ g_iw, const float* __restrict__ p_iw,
                            const float* __restrict__ g_cw, const float* __restrict__ p_cw,
                            const float* __restrict__ g_ib, const float* __restrict__ p_ib,
                            const float* __restrict__ g_bg, const float* __restrict__ g_bb,
                            const float* __restrict__ g_bm, const float* __restrict__ g_bv,
                            const float* __restrict__ p_bg, const float* __restrict__ p_bb,
                            const float* __restrict__ p_bm, const float* __restrict__ p_bv,
                            const float* __restrict__ g_cb, const float* __restrict__ p_cb,
                            _Float16* __restrict__ Wnode, _Float16* __restrict__ Wintro,
                            _Float16* __restrict__ Wcw,
                            float* __restrict__ sbn, float* __restrict__ tbn,
                            float* __restrict__ biasc, float* __restrict__ cbc)
{
    long long gid = (long long)blockIdx.x * blockDim.x + threadIdx.x;
    const long long NA = 2048LL * 1536, NBn = 1024LL * 1536, NC = 1024LL * 512;
    if (gid < NA) {
        int r = (int)(gid / 1536), k = (int)(gid % 1536);
        float v;
        if      (r < 512)  v = g_iw[(long long)r * 4608 + k];
        else if (r < 1024) v = g_iw[(long long)(r - 512) * 4608 + 1536 + k];
        else if (r < 1536) v = p_iw[(long long)(r - 1024) * 4608 + k];
        else               v = p_iw[(long long)(r - 1536) * 4608 + 1536 + k];
        Wnode[gid] = (_Float16)v;
    } else if (gid < NA + NBn) {
        long long t = gid - NA;
        int r = (int)(t / 1536), k = (int)(t % 1536);
        float v = (r < 512) ? g_iw[(long long)r * 4608 + 3072 + k]
                            : p_iw[(long long)(r - 512) * 4608 + 3072 + k];
        Wintro[t] = (_Float16)v;
    } else if (gid < NA + NBn + NC) {
        long long t = gid - NA - NBn;
        int r = (int)(t / 512), k = (int)(t % 512);
        float v = (r < 512) ? g_cw[(long long)r * 512 + k]
                            : p_cw[(long long)(r - 512) * 512 + k];
        Wcw[t] = (_Float16)v;
    } else if (gid < NA + NBn + NC + 1024) {
        int c = (int)(gid - (NA + NBn + NC));
        int u = c & 511;
        bool gg = (c < 512);
        float bg = gg ? g_bg[u] : p_bg[u];
        float bb = gg ? g_bb[u] : p_bb[u];
        float bm = gg ? g_bm[u] : p_bm[u];
        float bv = gg ? g_bv[u] : p_bv[u];
        float s = bg * rsqrtf(bv + 1e-5f);
        sbn[c]   = s;
        tbn[c]   = bb - bm * s;
        biasc[c] = gg ? g_ib[u] : p_ib[u];
        cbc[c]   = gg ? g_cb[u] : p_cb[u];
    }
}

// ---------------------------------------------------------------------------
// WMMA GEMM: C[M,N] = A[M,K](f16, row-major) x Bw[N,K](f16, row-major, i.e. B^T)
// 256-thread WG = 8 waves; wave tile 32x64: two M-subtiles share four B-frags
// -> per K-step: 4 A-loads + 8 B-loads for 8 WMMAs (1.5 loads/wmma).
// WG tile 64x256. Fragment layouts per CDNA5 ISA 7.12.2.
//   MODE 0: plain store (node projections, Ntot=2048)
//   MODE 1: + bias + Anode[i] + Anode[j]; store H (f32) and D=f16(relu(BN)) — intro
//   MODE 2: block-diagonal K (A column-half selected by n_base); H += acc + cb — chorus
// ---------------------------------------------------------------------------
template <int MODE>
__global__ __launch_bounds__(256)
void gemm_kernel(const _Float16* __restrict__ A, const _Float16* __restrict__ Bw,
                 int K, int lda, int ldb, int Ntot,
                 float* __restrict__ Cout, _Float16* __restrict__ Dout,
                 const float* __restrict__ Anode,
                 const float* __restrict__ sbn, const float* __restrict__ tbn,
                 const float* __restrict__ biasc, const float* __restrict__ cbc)
{
    int lane = threadIdx.x & 31;
    int w    = threadIdx.x >> 5;
    int m_base = blockIdx.x * 64 + (w & 1) * 32;   // wave covers 32 rows
    int n_base = blockIdx.y * 256 + (w >> 1) * 64; // and 64 cols

    int khalf8 = (lane >> 4) * 8;     // A-frag: lanes 0-15 K{0..7,16..23}; 16-31 K{8..15,24..31}
    int bcol0  = n_base + (lane & 15);
    int kh16   = (lane >> 4) * 8;     // B-frag: same K split as A-frag

    const _Float16* Aptr0 = A + (long long)(m_base + (lane & 15)) * lda;
    if (MODE == 2 && n_base >= 512) Aptr0 += 512;   // second diagonal block (pr net)
    const _Float16* Aptr1 = Aptr0 + (long long)16 * lda;

    v8f acc[2][4] = {};
    for (int k = 0; k < K; k += 32) {
        // prefetch next K-chunk of the A stream (lowers to global_prefetch_b8)
        __builtin_prefetch(Aptr0 + k + 128, 0, 1);
        __builtin_prefetch(Aptr1 + k + 128, 0, 1);

        v8h a0lo = *(const v8h*)(Aptr0 + k + khalf8);
        v8h a0hi = *(const v8h*)(Aptr0 + k + 16 + khalf8);
        v8h a1lo = *(const v8h*)(Aptr1 + k + khalf8);
        v8h a1hi = *(const v8h*)(Aptr1 + k + 16 + khalf8);
        v16h a0 = __builtin_shufflevector(a0lo, a0hi, 0, 1, 2, 3, 4, 5, 6, 7,
                                                      8, 9, 10, 11, 12, 13, 14, 15);
        v16h a1 = __builtin_shufflevector(a1lo, a1hi, 0, 1, 2, 3, 4, 5, 6, 7,
                                                      8, 9, 10, 11, 12, 13, 14, 15);
#pragma unroll
        for (int s = 0; s < 4; ++s) {
            const _Float16* bp = Bw + (long long)(bcol0 + s * 16) * ldb + k + kh16;
            v16h bfrag = __builtin_shufflevector(*(const v8h*)bp, *(const v8h*)(bp + 16),
                                                 0, 1, 2, 3, 4, 5, 6, 7, 8, 9, 10, 11, 12, 13, 14, 15);
            acc[0][s] = __builtin_amdgcn_wmma_f32_16x16x32_f16(
                false, a0, false, bfrag, (short)0, acc[0][s], false, false);
            acc[1][s] = __builtin_amdgcn_wmma_f32_16x16x32_f16(
                false, a1, false, bfrag, (short)0, acc[1][s], false, false);
        }
    }

#pragma unroll
    for (int s = 0; s < 4; ++s) {
        int c = n_base + s * 16 + (lane & 15);
        float bia = 0.f, sc = 0.f, sh = 0.f, cbv = 0.f;
        if (MODE == 1) { bia = biasc[c]; sc = sbn[c]; sh = tbn[c]; }
        if (MODE == 2) { cbv = cbc[c]; }
#pragma unroll
        for (int mi = 0; mi < 2; ++mi) {
#pragma unroll
            for (int v = 0; v < 8; ++v) {
                int m = m_base + mi * 16 + v + (lane >> 4) * 8;
                float val = acc[mi][s][v];
                if (MODE == 0) {
                    Cout[(long long)m * Ntot + c] = val;
                } else if (MODE == 1) {
                    int b = m >> 10, e = m & 1023, i = e >> 5, j = e & 31;
                    int ca  = (c < 512) ? c : c + 512;        // W1 block column in Anode
                    int cb2 = (c < 512) ? c + 512 : c + 1024; // W2 block column in Anode
                    float h = val + bia
                            + Anode[(long long)(b * 32 + i) * 2048 + ca]
                            + Anode[(long long)(b * 32 + j) * 2048 + cb2];
                    long long idx = (long long)m * 1024 + c;
                    Cout[idx] = h;                              // residual source
                    float d = h * sc + sh;                      // BN folded
                    Dout[idx] = (_Float16)fmaxf(d, 0.f);        // ReLU
                } else {
                    long long idx = (long long)m * 1024 + c;
                    Cout[idx] = Cout[idx] + val + cbv;          // x + chorus(d)
                }
            }
        }
    }
}

// ---------------------------------------------------------------------------
// Kernel 4: outro heads. One block (64 threads) per edge.
// gate = sigmoid(g_ow . Rg + g_ob); pr_o = p_ow[o] . Rp + p_ob[o];
// out = gate * pr / max(||pr||, 1e-12) * mask.
// ---------------------------------------------------------------------------
__global__ __launch_bounds__(64)
void outro_kernel(const float* __restrict__ R,
                  const float* __restrict__ g_ow, const float* __restrict__ g_ob,
                  const float* __restrict__ p_ow, const float* __restrict__ p_ob,
                  const int* __restrict__ nvec, float* __restrict__ out)
{
    __shared__ float spr[64];
    __shared__ float sgate;
    int m = blockIdx.x;             // edge row 0..8191
    int o = threadIdx.x;            // out channel 0..63
    const float* Rg = R + (long long)m * 1024;
    const float* Rp = Rg + 512;
    const float* wo = p_ow + o * 512;

    float accp = p_ob[o];
    for (int c = 0; c < 512; ++c) accp += Rp[c] * wo[c];

    if (o == 0) {
        float g = g_ob[0];
        for (int c = 0; c < 512; ++c) g += Rg[c] * g_ow[c];
        sgate = 1.f / (1.f + expf(-g));
    }
    spr[o] = accp;
    __syncthreads();

    float ss = 0.f;
    for (int t = 0; t < 64; ++t) ss += spr[t] * spr[t];
    float norm = fmaxf(sqrtf(ss), 1e-12f);

    int b = m >> 10, e = m & 1023, i = e >> 5, j = e & 31;
    float mask = (i < nvec[b] && j < nvec[b]) ? 1.f : 0.f;
    out[(((long long)b * 64 + o) * 32 + i) * 32 + j] = sgate * accp / norm * mask;
}

// ---------------------------------------------------------------------------
extern "C" void kernel_launch(void* const* d_in, const int* in_sizes, int n_in,
                              void* d_out, int out_size, void* d_ws, size_t ws_size,
                              hipStream_t stream)
{
    const float* f0 = (const float*)d_in[0];
    const float* f1 = (const float*)d_in[1];
    const float* f2 = (const float*)d_in[2];
    const float* f3 = (const float*)d_in[3];
    const float* f4 = (const float*)d_in[4];
    const float* f5 = (const float*)d_in[5];
    const float* P    = (const float*)d_in[6];
    const int*   nvec = (const int*)d_in[7];
    const float* g_iw = (const float*)d_in[8];
    const float* g_ib = (const float*)d_in[9];
    const float* g_bg = (const float*)d_in[10];
    const float* g_bb = (const float*)d_in[11];
    const float* g_bm = (const float*)d_in[12];
    const float* g_bv = (const float*)d_in[13];
    const float* g_cw = (const float*)d_in[14];
    const float* g_cb = (const float*)d_in[15];
    const float* g_ow = (const float*)d_in[16];
    const float* g_ob = (const float*)d_in[17];
    const float* p_iw = (const float*)d_in[18];
    const float* p_ib = (const float*)d_in[19];
    const float* p_bg = (const float*)d_in[20];
    const float* p_bb = (const float*)d_in[21];
    const float* p_bm = (const float*)d_in[22];
    const float* p_bv = (const float*)d_in[23];
    const float* p_cw = (const float*)d_in[24];
    const float* p_cb = (const float*)d_in[25];
    const float* p_ow = (const float*)d_in[26];
    const float* p_ob = (const float*)d_in[27];

    // workspace carve (256B aligned)
    char* ws = (char*)d_ws;
    auto carve = [&](size_t bytes) {
        char* p = ws;
        ws += (bytes + 255) & ~(size_t)255;
        return p;
    };
    _Float16* Fn     = (_Float16*)carve((size_t)NNODE * LATC * 2);   //  0.8 MB
    _Float16* Ee     = (_Float16*)carve((size_t)MROWS * LATC * 2);   // 25.2 MB
    _Float16* Wnode  = (_Float16*)carve((size_t)2048 * 1536 * 2);    //  6.3 MB
    _Float16* Wintro = (_Float16*)carve((size_t)1024 * 1536 * 2);    //  3.1 MB
    _Float16* Wcw    = (_Float16*)carve((size_t)1024 * 512 * 2);     //  1.0 MB
    float*    Anode  = (float*)carve((size_t)NNODE * 2048 * 4);      //  2.1 MB
    float*    H      = (float*)carve((size_t)MROWS * 1024 * 4);      // 33.6 MB (also R, in place)
    _Float16* D      = (_Float16*)carve((size_t)MROWS * 1024 * 2);   // 16.8 MB
    float*    sbn    = (float*)carve(1024 * 4);
    float*    tbn    = (float*)carve(1024 * 4);
    float*    biasc  = (float*)carve(1024 * 4);
    float*    cbc    = (float*)carve(1024 * 4);

    // 1) grid-sample nodes + edge midpoints
    align_kernel<<<NNODE + MROWS, 256, 0, stream>>>(f0, f1, f2, f3, f4, f5, P, Fn, Ee);

    // 2) weight repack + BN fold
    long long preptot = 2048LL * 1536 + 1024LL * 1536 + 1024LL * 512 + 1024;
    prep_kernel<<<(int)((preptot + 255) / 256), 256, 0, stream>>>(
        g_iw, p_iw, g_cw, p_cw, g_ib, p_ib,
        g_bg, g_bb, g_bm, g_bv, p_bg, p_bb, p_bm, p_bv, g_cb, p_cb,
        Wnode, Wintro, Wcw, sbn, tbn, biasc, cbc);

    // 3) node projections: Anode[256,2048] = Fn[256,1536] x Wnode^T
    dim3 g0(NNODE / 64, 2048 / 256);
    gemm_kernel<0><<<g0, 256, 0, stream>>>(Fn, Wnode, 1536, 1536, 1536, 2048,
                                           Anode, nullptr, nullptr,
                                           nullptr, nullptr, nullptr, nullptr);

    // 4) intro: H = E x W3^T + bias + Anode[i] + Anode[j]; D = relu(BN(H)) in f16
    dim3 g1(MROWS / 64, 1024 / 256);
    gemm_kernel<1><<<g1, 256, 0, stream>>>(Ee, Wintro, 1536, 1536, 1536, 1024,
                                           H, D, Anode, sbn, tbn, biasc, nullptr);

    // 5) chorus (block-diagonal K=512), residual in place: H += D x Wcw^T + cb
    gemm_kernel<2><<<g1, 256, 0, stream>>>(D, Wcw, 512, 1024, 512, 1024,
                                           H, nullptr, nullptr,
                                           nullptr, nullptr, nullptr, cbc);

    // 6) outro heads + sigmoid gate + L2 normalize + mask
    outro_kernel<<<MROWS, 64, 0, stream>>>(H, g_ow, g_ob, p_ow, p_ob, nvec,
                                           (float*)d_out);
}
